// RAEViT_37941741093299
// MI455X (gfx1250) — compile-verified, matrix operand order checked by reading, not run on hardware
//
#include <hip/hip_runtime.h>

typedef __bf16 bf16;
typedef __bf16 v16bf __attribute__((ext_vector_type(16)));
typedef float  v8f   __attribute__((ext_vector_type(8)));

#define WMMA_BF16(A_, B_, Cacc_) \
  __builtin_amdgcn_wmma_f32_16x16x32_bf16(false, (A_), false, (B_), (short)0, (Cacc_), false, false)

#define CDIM  768
#define NTOK  1729      // 1 cls + 12^3 patches
#define NP    1760      // padded to 55*32 (WMMA K pairs)
#define BATCH 2
#define NH    8
#define HD    96
#define NPATCH 1728
#define BSTRIDE 40      // bf16 elems per LDS B-strip row (80B, 16B-aligned, conflict-spread)

// ---------------------------------------------------------------------------
// WMMA fragment loader: A-style and B-style fragments share the same per-lane
// K striping (ISA 7.12.2, 16-bit 16x32 layout):
//   lane L holds row/col (L&15); element e maps to K = e + (e>=8?8:0) + (L>=16?8:0)
// which is two contiguous 8-element bf16 runs at k0+{0,16}+(L>=16?8:0).
// ---------------------------------------------------------------------------
__device__ inline v16bf load_frag(const bf16* base, int ld, int row, int k0, int lane) {
  const bf16* p = base + (size_t)(row + (lane & 15)) * ld + k0 + ((lane >> 4) << 3);
  v16bf f;
#pragma unroll
  for (int e = 0; e < 8; ++e) { f[e] = p[e]; f[e + 8] = p[e + 16]; }
  return f;
}

// ---------------------------------------------------------------------------
// generic elementwise f32 -> bf16 pack (weights)
// ---------------------------------------------------------------------------
__global__ __launch_bounds__(256) void f2bf_kernel(const float* __restrict__ in,
                                                   bf16* __restrict__ out, int n) {
  int i = blockIdx.x * 256 + threadIdx.x;
  if (i < n) out[i] = (bf16)in[i];
}

// ---------------------------------------------------------------------------
// patchify: smri (B,1,96,96,96) -> Xp bf16 [B*1728, 512], within-patch i*64+j*8+k
// ---------------------------------------------------------------------------
__global__ __launch_bounds__(256) void patchify_kernel(const float* __restrict__ smri,
                                                       bf16* __restrict__ xp) {
  int row = blockIdx.x;                       // b*1728 + patch
  int b = row / NPATCH, pidx = row % NPATCH;
  int px = pidx / 144, py = (pidx / 12) % 12, pz = pidx % 12;
  for (int c = threadIdx.x; c < 512; c += 256) {
    int i = c >> 6, j = (c >> 3) & 7, kk = c & 7;
    size_t src = (((size_t)b * 96 + px * 8 + i) * 96 + (py * 8 + j)) * 96 + (pz * 8 + kk);
    xp[(size_t)row * 512 + c] = (bf16)smri[src];
  }
}

// ---------------------------------------------------------------------------
// bf16 WMMA GEMM, one wave per 16(M) x 64(N) strip:  C = A[M,K] @ B[N,K]^T + bias
//  - A fragment: direct global b128 loads (reused across 4 N-tiles)
//  - B strip (64 rows x 32 K = 4KB): staged to LDS with ASYNCcnt-tracked
//    global_load_async_to_lds_b128 (8 x 512B per K-step), then DS fragment reads.
//  - s_wait_dscnt 0 guards the LDS overwrite (ASYNCcnt and DScnt are unordered).
// K multiple of 32, M multiple of 16, N multiple of 64. outBf: 1=bf16, 0=f32
// ---------------------------------------------------------------------------
__global__ __launch_bounds__(32) void gemm_bf16_kernel(const bf16* __restrict__ A,
                                                       const bf16* __restrict__ B,
                                                       const float* __restrict__ bias,
                                                       void* __restrict__ Cptr,
                                                       int M, int N, int K, int outBf) {
  __shared__ bf16 bs[64 * BSTRIDE];
  int lane = threadIdx.x;
  int tilesN = N >> 6;
  int tm = blockIdx.x / tilesN, tn = blockIdx.x % tilesN;
  int n0 = tn << 6;
  v8f acc[4] = {};

  int srow = lane >> 2;                 // 0..7 : B row within an 8-row async group
  int schunk = (lane & 3) << 3;         // 0,8,16,24 : bf16 offset of this lane's 16B
  unsigned lds0 = (unsigned)(uintptr_t)(&bs[0]);

  for (int k0 = 0; k0 < K; k0 += 32) {
    // all DS reads of the previous strip must retire before we overwrite it
    asm volatile("s_wait_dscnt 0x0" ::: "memory");
#pragma unroll
    for (int op = 0; op < 8; ++op) {
      int r = op * 8 + srow;
      const bf16* g = B + (size_t)(n0 + r) * K + k0 + schunk;
      unsigned l = lds0 + (unsigned)((r * BSTRIDE + schunk) * 2);
      asm volatile("global_load_async_to_lds_b128 %0, %1, off"
                   :: "v"(l), "v"(g) : "memory");
    }
    v16bf a = load_frag(A, K, tm * 16, k0, lane);
    if (k0 + 32 < K)
      __builtin_prefetch((const void*)(A + (size_t)(tm * 16 + (lane & 15)) * K + k0 + 32), 0, 0);
    asm volatile("s_wait_asynccnt 0x0" ::: "memory");
    v16bf b0 = load_frag(bs, BSTRIDE, 0,  0, lane);
    v16bf b1 = load_frag(bs, BSTRIDE, 16, 0, lane);
    v16bf b2 = load_frag(bs, BSTRIDE, 32, 0, lane);
    v16bf b3 = load_frag(bs, BSTRIDE, 48, 0, lane);
    acc[0] = WMMA_BF16(a, b0, acc[0]);
    acc[1] = WMMA_BF16(a, b1, acc[1]);
    acc[2] = WMMA_BF16(a, b2, acc[2]);
    acc[3] = WMMA_BF16(a, b3, acc[3]);
  }

  int mbase = tm * 16 + ((lane >> 4) << 3);
#pragma unroll
  for (int t = 0; t < 4; ++t) {
    int nn = n0 + t * 16 + (lane & 15);
    float bv = bias ? bias[nn] : 0.f;
    if (outBf) {
      bf16* Cb = (bf16*)Cptr;
#pragma unroll
      for (int r = 0; r < 8; ++r) Cb[(size_t)(mbase + r) * N + nn] = (bf16)(acc[t][r] + bv);
    } else {
      float* Cf = (float*)Cptr;
#pragma unroll
      for (int r = 0; r < 8; ++r) Cf[(size_t)(mbase + r) * N + nn] = acc[t][r] + bv;
    }
  }
}

// ---------------------------------------------------------------------------
// block reductions (256 threads)
// ---------------------------------------------------------------------------
__device__ inline float block_sum(float v, float* red) {
  int tid = threadIdx.x;
  red[tid] = v; __syncthreads();
  for (int s = 128; s > 0; s >>= 1) { if (tid < s) red[tid] += red[tid + s]; __syncthreads(); }
  float r = red[0]; __syncthreads(); return r;
}
__device__ inline float block_max(float v, float* red) {
  int tid = threadIdx.x;
  red[tid] = v; __syncthreads();
  for (int s = 128; s > 0; s >>= 1) { if (tid < s) red[tid] = fmaxf(red[tid], red[tid + s]); __syncthreads(); }
  float r = red[0]; __syncthreads(); return r;
}

// ---------------------------------------------------------------------------
// LayerNorm(patch-embed) + cls token + pos embed -> xb bf16 (B, NP, C), pad rows zero
// ---------------------------------------------------------------------------
__global__ __launch_bounds__(256) void ln_assemble_kernel(const float* __restrict__ pe,
                                                          const float* __restrict__ g,
                                                          const float* __restrict__ bt,
                                                          const float* __restrict__ cls,
                                                          const float* __restrict__ pos,
                                                          bf16* __restrict__ xb) {
  __shared__ float red[256];
  int row = blockIdx.x;                        // b*NP + t
  int b = row / NP, t = row % NP;
  bf16* out = xb + (size_t)row * CDIM;
  if (t >= NTOK) { for (int c = threadIdx.x; c < CDIM; c += 256) out[c] = (bf16)0.f; return; }
  if (t == 0)    { for (int c = threadIdx.x; c < CDIM; c += 256) out[c] = (bf16)(cls[c] + pos[c]); return; }
  const float* x = pe + ((size_t)b * NPATCH + (t - 1)) * CDIM;
  float s = 0.f;
  for (int c = threadIdx.x; c < CDIM; c += 256) s += x[c];
  float mean = block_sum(s, red) / (float)CDIM;
  float v = 0.f;
  for (int c = threadIdx.x; c < CDIM; c += 256) { float d = x[c] - mean; v += d * d; }
  float rstd = rsqrtf(block_sum(v, red) / (float)CDIM + 1e-5f);
  const float* pr = pos + (size_t)t * CDIM;
  for (int c = threadIdx.x; c < CDIM; c += 256)
    out[c] = (bf16)((x[c] - mean) * rstd * g[c] + bt[c] + pr[c]);
}

// ---------------------------------------------------------------------------
// Regional attention, flash-style. One wave per (b, h, 16-query tile).
// Scores via WMMA (QK^T), online softmax with 16-lane shfl reductions,
// P transposed D-layout -> A-layout through LDS, P@V via WMMA.
// Region bias omitted: constant per query over the softmax axis => no-op.
// ---------------------------------------------------------------------------
__global__ __launch_bounds__(32) void region_attn_kernel(const bf16* __restrict__ q,
                                                         const bf16* __restrict__ k,
                                                         const bf16* __restrict__ v,
                                                         bf16* __restrict__ xo) {
  __shared__ float ps[16][33];
  int lane = threadIdx.x;
  int qb = blockIdx.x % (NP / 16);
  int h  = (blockIdx.x / (NP / 16)) % NH;
  int b  =  blockIdx.x / ((NP / 16) * NH);
  const bf16* qbp = q + (size_t)b * NP * CDIM;
  const bf16* kbp = k + (size_t)b * NP * CDIM;
  const bf16* vbp = v + (size_t)b * NP * CDIM;

  v16bf aq[3];
#pragma unroll
  for (int c = 0; c < 3; ++c) aq[c] = load_frag(qbp, CDIM, qb * 16, h * HD + 32 * c, lane);

  v8f o[6] = {};
  float mi[8], li[8];
#pragma unroll
  for (int r = 0; r < 8; ++r) { mi[r] = -1e30f; li[r] = 0.f; }

  const float scale = 0.1020620726f;  // 1/sqrt(96)
  int base = (lane >> 4) << 3;
  int klo_lane = lane & 15;

  for (int p = 0; p < NP / 32; ++p) {
    v8f slo = {}, shi = {};
#pragma unroll
    for (int c = 0; c < 3; ++c) {
      v16bf bl = load_frag(kbp, CDIM, p * 32,      h * HD + 32 * c, lane);
      v16bf bh = load_frag(kbp, CDIM, p * 32 + 16, h * HD + 32 * c, lane);
      slo = WMMA_BF16(aq[c], bl, slo);
      shi = WMMA_BF16(aq[c], bh, shi);
    }
    bool mlo = (p * 32 + klo_lane)      >= NTOK;
    bool mhi = (p * 32 + klo_lane + 16) >= NTOK;
#pragma unroll
    for (int r = 0; r < 8; ++r) {
      float a  = mlo ? -1e30f : slo[r] * scale;
      float bb = mhi ? -1e30f : shi[r] * scale;
      float mx = fmaxf(a, bb);
#pragma unroll
      for (int off = 1; off < 16; off <<= 1) mx = fmaxf(mx, __shfl_xor(mx, off));
      float mn   = fmaxf(mi[r], mx);
      float corr = __expf(mi[r] - mn);
      float pl = __expf(a - mn), ph = __expf(bb - mn);
      float rs = pl + ph;
#pragma unroll
      for (int off = 1; off < 16; off <<= 1) rs += __shfl_xor(rs, off);
      li[r] = li[r] * corr + rs;
      mi[r] = mn;
#pragma unroll
      for (int t = 0; t < 6; ++t) o[t][r] *= corr;
      ps[base + r][klo_lane]      = pl;
      ps[base + r][klo_lane + 16] = ph;
    }
    __syncthreads();
    // D-layout -> A-layout transpose of P through LDS
    v16bf ap;
#pragma unroll
    for (int e = 0; e < 8; ++e) {
      ap[e]     = (bf16)ps[klo_lane][base + e];
      ap[e + 8] = (bf16)ps[klo_lane][base + 16 + e];
    }
#pragma unroll
    for (int t = 0; t < 6; ++t) {
      const bf16* vp = vbp + (size_t)(p * 32) * CDIM + h * HD + t * 16 + klo_lane;
      v16bf bv;
#pragma unroll
      for (int e = 0; e < 8; ++e) {
        bv[e]     = vp[(size_t)(base + e) * CDIM];
        bv[e + 8] = vp[(size_t)(base + 16 + e) * CDIM];
      }
      o[t] = WMMA_BF16(ap, bv, o[t]);
    }
    __syncthreads();
  }
#pragma unroll
  for (int r = 0; r < 8; ++r) {
    float inv = 1.f / li[r];
    int tok = qb * 16 + base + r;
    bf16* orow = xo + ((size_t)b * NP + tok) * CDIM + h * HD + klo_lane;
#pragma unroll
    for (int t = 0; t < 6; ++t) orow[t * 16] = (bf16)(o[t][r] * inv);
  }
}

// ---------------------------------------------------------------------------
// Branch attention, CLS query only (all other rows are dead code).
// qkv: (B, NP, 3C) bf16 from the in-projection GEMM. out: (B, C) f32.
// ---------------------------------------------------------------------------
__global__ __launch_bounds__(256) void branch_attn_kernel(const bf16* __restrict__ qkv,
                                                          float* __restrict__ out) {
  __shared__ float q0s[HD];
  __shared__ float sc[NTOK];
  __shared__ float red[256];
  int b = blockIdx.x >> 3, h = blockIdx.x & 7;
  int tid = threadIdx.x;
  const bf16* basep = qkv + (size_t)b * NP * (3 * CDIM);
  if (tid < HD) q0s[tid] = (float)basep[h * HD + tid];
  __syncthreads();
  const float scale = 0.1020620726f;
  for (int key = tid; key < NTOK; key += 256) {
    const bf16* kp = basep + (size_t)key * (3 * CDIM) + CDIM + h * HD;
    float s = 0.f;
#pragma unroll 8
    for (int j = 0; j < HD; ++j) s += q0s[j] * (float)kp[j];
    sc[key] = s * scale;
  }
  __syncthreads();
  float m = -1e30f;
  for (int key = tid; key < NTOK; key += 256) m = fmaxf(m, sc[key]);
  m = block_max(m, red);
  float l = 0.f;
  for (int key = tid; key < NTOK; key += 256) l += __expf(sc[key] - m);
  l = block_sum(l, red);
  float invl = 1.f / l;
  for (int key = tid; key < NTOK; key += 256) sc[key] = __expf(sc[key] - m) * invl;
  __syncthreads();
  if (tid < HD) {
    float acc = 0.f;
    const bf16* vp = basep + 2 * CDIM + h * HD + tid;
    for (int key = 0; key < NTOK; ++key) acc += sc[key] * (float)vp[(size_t)key * (3 * CDIM)];
    out[(size_t)b * CDIM + h * HD + tid] = acc;
  }
}

// ---------------------------------------------------------------------------
// CLS out-projections + weighted combine + LayerNorm + head. One block per batch.
// ---------------------------------------------------------------------------
__global__ __launch_bounds__(256) void head_kernel(const float* __restrict__ ol,
                                                   const float* __restrict__ og,
                                                   const float* __restrict__ low,
                                                   const float* __restrict__ lob,
                                                   const float* __restrict__ gow,
                                                   const float* __restrict__ gob,
                                                   const float* __restrict__ wl,
                                                   const float* __restrict__ wg,
                                                   const float* __restrict__ lng,
                                                   const float* __restrict__ lnb,
                                                   const float* __restrict__ hw,
                                                   const float* __restrict__ hb,
                                                   float* __restrict__ out) {
  __shared__ float xs[CDIM];
  __shared__ float red[256];
  int b = blockIdx.x, tid = threadIdx.x;
  const float* olb = ol + (size_t)b * CDIM;
  const float* ogb = og + (size_t)b * CDIM;
  float wlv = wl[0], wgv = wg[0];
  for (int n = tid; n < CDIM; n += 256) {
    float al = lob[n], ag = gob[n];
    const float* lr = low + (size_t)n * CDIM;
    const float* gr = gow + (size_t)n * CDIM;
    for (int c = 0; c < CDIM; ++c) { al += olb[c] * lr[c]; ag += ogb[c] * gr[c]; }
    xs[n] = wlv * al + wgv * ag;
  }
  __syncthreads();
  float s = 0.f;
  for (int n = tid; n < CDIM; n += 256) s += xs[n];
  float mean = block_sum(s, red) / (float)CDIM;
  float v = 0.f;
  for (int n = tid; n < CDIM; n += 256) { float d = xs[n] - mean; v += d * d; }
  float rstd = rsqrtf(block_sum(v, red) / (float)CDIM + 1e-5f);
  for (int n = tid; n < CDIM; n += 256) xs[n] = (xs[n] - mean) * rstd * lng[n] + lnb[n];
  __syncthreads();
  for (int cl = 0; cl < 3; ++cl) {
    float p = 0.f;
    for (int n = tid; n < CDIM; n += 256) p += xs[n] * hw[(size_t)cl * CDIM + n];
    float tot = block_sum(p, red);
    if (tid == 0) out[b * 3 + cl] = tot + hb[cl];
  }
}

// ---------------------------------------------------------------------------
extern "C" void kernel_launch(void* const* d_in, const int* in_sizes, int n_in,
                              void* d_out, int out_size, void* d_ws, size_t ws_size,
                              hipStream_t stream) {
  (void)in_sizes; (void)n_in; (void)out_size; (void)ws_size;
  const float* smri     = (const float*)d_in[0];
  // d_in[1] region_mask, d_in[12] region_weight: softmax-shift-invariant -> dead
  const float* conv_w   = (const float*)d_in[2];
  const float* conv_b   = (const float*)d_in[3];
  const float* pe_ln_g  = (const float*)d_in[4];
  const float* pe_ln_b  = (const float*)d_in[5];
  const float* q_w      = (const float*)d_in[6];
  const float* q_b      = (const float*)d_in[7];
  const float* k_w      = (const float*)d_in[8];
  const float* k_b      = (const float*)d_in[9];
  const float* v_w      = (const float*)d_in[10];
  const float* v_b      = (const float*)d_in[11];
  const float* loc_in_w = (const float*)d_in[13];
  const float* loc_in_b = (const float*)d_in[14];
  const float* loc_out_w= (const float*)d_in[15];
  const float* loc_out_b= (const float*)d_in[16];
  const float* glob_in_w = (const float*)d_in[17];
  const float* glob_in_b = (const float*)d_in[18];
  const float* glob_out_w= (const float*)d_in[19];
  const float* glob_out_b= (const float*)d_in[20];
  const float* w_local  = (const float*)d_in[21];
  const float* w_global = (const float*)d_in[22];
  const float* cls_tok  = (const float*)d_in[23];
  const float* pos      = (const float*)d_in[24];
  const float* head_ln_g= (const float*)d_in[25];
  const float* head_ln_b= (const float*)d_in[26];
  const float* head_w   = (const float*)d_in[27];
  const float* head_b   = (const float*)d_in[28];

  char* ws = (char*)d_ws;
  size_t off = 0;
  auto alloc = [&](size_t bytes) { size_t o = off; off += (bytes + 255) & ~(size_t)255; return o; };

  const int M1 = BATCH * NPATCH;   // 3456
  const int Mx = BATCH * NP;       // 3520

  bf16* wc  = (bf16*)(ws + alloc((size_t)CDIM * 512 * 2));
  bf16* wq  = (bf16*)(ws + alloc((size_t)CDIM * CDIM * 2));
  bf16* wk  = (bf16*)(ws + alloc((size_t)CDIM * CDIM * 2));
  bf16* wv  = (bf16*)(ws + alloc((size_t)CDIM * CDIM * 2));
  bf16* wli = (bf16*)(ws + alloc((size_t)3 * CDIM * CDIM * 2));
  bf16* wgi = (bf16*)(ws + alloc((size_t)3 * CDIM * CDIM * 2));
  bf16* xp  = (bf16*)(ws + alloc((size_t)M1 * 512 * 2));
  float* pe = (float*)(ws + alloc((size_t)M1 * CDIM * 4));
  bf16* xb  = (bf16*)(ws + alloc((size_t)Mx * CDIM * 2));
  bf16* qr  = (bf16*)(ws + alloc((size_t)Mx * CDIM * 2));
  bf16* kr  = (bf16*)(ws + alloc((size_t)Mx * CDIM * 2));
  bf16* vr  = (bf16*)(ws + alloc((size_t)Mx * CDIM * 2));
  bf16* xa  = (bf16*)(ws + alloc((size_t)Mx * CDIM * 2));
  bf16* ql  = (bf16*)(ws + alloc((size_t)Mx * 3 * CDIM * 2));
  bf16* qg  = (bf16*)(ws + alloc((size_t)Mx * 3 * CDIM * 2));
  float* ol = (float*)(ws + alloc((size_t)BATCH * CDIM * 4));
  float* og = (float*)(ws + alloc((size_t)BATCH * CDIM * 4));

  // weight packs f32 -> bf16
  auto pack = [&](const float* src, bf16* dst, int n) {
    f2bf_kernel<<<(n + 255) / 256, 256, 0, stream>>>(src, dst, n);
  };
  pack(conv_w,    wc,  CDIM * 512);
  pack(q_w,       wq,  CDIM * CDIM);
  pack(k_w,       wk,  CDIM * CDIM);
  pack(v_w,       wv,  CDIM * CDIM);
  pack(loc_in_w,  wli, 3 * CDIM * CDIM);
  pack(glob_in_w, wgi, 3 * CDIM * CDIM);

  // patch embed
  patchify_kernel<<<M1, 256, 0, stream>>>(smri, xp);
  gemm_bf16_kernel<<<(M1 / 16) * (CDIM / 64), 32, 0, stream>>>(
      xp, wc, conv_b, pe, M1, CDIM, 512, 0);
  ln_assemble_kernel<<<Mx, 256, 0, stream>>>(pe, pe_ln_g, pe_ln_b, cls_tok, pos, xb);

  // regional q/k/v projections
  gemm_bf16_kernel<<<(Mx / 16) * (CDIM / 64), 32, 0, stream>>>(xb, wq, q_b, qr, Mx, CDIM, CDIM, 1);
  gemm_bf16_kernel<<<(Mx / 16) * (CDIM / 64), 32, 0, stream>>>(xb, wk, k_b, kr, Mx, CDIM, CDIM, 1);
  gemm_bf16_kernel<<<(Mx / 16) * (CDIM / 64), 32, 0, stream>>>(xb, wv, v_b, vr, Mx, CDIM, CDIM, 1);

  // flash regional attention
  region_attn_kernel<<<BATCH * NH * (NP / 16), 32, 0, stream>>>(qr, kr, vr, xa);

  // branch in-projections (full), then CLS-only attention per branch
  gemm_bf16_kernel<<<(Mx / 16) * (3 * CDIM / 64), 32, 0, stream>>>(
      xa, wli, loc_in_b, ql, Mx, 3 * CDIM, CDIM, 1);
  gemm_bf16_kernel<<<(Mx / 16) * (3 * CDIM / 64), 32, 0, stream>>>(
      xa, wgi, glob_in_b, qg, Mx, 3 * CDIM, CDIM, 1);
  branch_attn_kernel<<<BATCH * NH, 256, 0, stream>>>(ql, ol);
  branch_attn_kernel<<<BATCH * NH, 256, 0, stream>>>(qg, og);

  // out-proj (CLS rows) + combine + LN + head
  head_kernel<<<BATCH, 256, 0, stream>>>(ol, og, loc_out_w, loc_out_b, glob_out_w, glob_out_b,
                                         w_local, w_global, head_ln_g, head_ln_b,
                                         head_w, head_b, (float*)d_out);
}
